// VectorQuantizedVAE_26594437497067
// MI455X (gfx1250) — matrix operand order repaired and checked by reading, not compiled, not run on hardware
//
#include <hip/hip_runtime.h>
#include <hip/hip_bf16.h>
#include <stddef.h>
#include <stdint.h>

// ---------------------------------------------------------------------------
// VQ-VAE forward on gfx1250: convs lowered to im2col + bf16 WMMA GEMM.
// GEMM: 128x128 block tile, 8 waves x (16M x 128N), K-step 32.
// Software-pipelined: double-buffered LDS tiles; next A tile held in VGPRs
// (b64 loads, packed bf16 b32 commits) and next B tile in flight via
// global_load_async_to_lds_b128 (ASYNCcnt) while the current tile runs
// 8x v_wmma_f32_16x16x32_bf16. FLIP (ConvTranspose tap reversal) is a
// template parameter so the A-staging pack order is branch/select-free.
// ---------------------------------------------------------------------------

typedef __bf16 bf16;
typedef __attribute__((ext_vector_type(16))) __bf16 v16bf;
typedef __attribute__((ext_vector_type(8)))  __bf16 v8bf;
typedef __attribute__((ext_vector_type(8)))  float  v8f;
typedef __attribute__((ext_vector_type(2)))  float  v2f;

#define TM 128
#define TN 128
#define TK 32
#define APITCH 40   // bf16 elems per LDS A row (80 B, keeps 16B-aligned chunks)

// ======================= im2col (f32 -> bf16, N-major) =====================
__global__ void vqvae_im2col_kernel(const float* __restrict__ x, bf16* __restrict__ Bmt,
                                    int Bsz, int Cin, int Lin, int ktap,
                                    int stride, int pad, int lhs_dil, int Lout) {
    int Ktot = Cin * ktap;
    size_t idx = (size_t)blockIdx.x * blockDim.x + threadIdx.x;
    size_t total = (size_t)Ktot * Bsz * Lout;
    if (idx >= total) return;
    int k = (int)(idx % Ktot);
    int n = (int)(idx / Ktot);
    int ci = k / ktap, t = k - ci * ktap;
    int b = n / Lout, lo = n - b * Lout;
    int p = lo * stride + t - pad;
    float v = 0.0f;
    if (p >= 0) {
        if (lhs_dil == 1) {
            if (p < Lin) v = x[((size_t)b * Cin + ci) * Lin + p];
        } else if ((p % lhs_dil) == 0) {
            int s = p / lhs_dil;
            if (s < Lin) v = x[((size_t)b * Cin + ci) * Lin + s];
        }
    }
    Bmt[idx] = (bf16)v;
}

// ======================= WMMA GEMM =========================================
// A column offset for (m, gk): gk (no flip) | gk^3 (flip, ktap==4), since
// ci*ktap + t == gk and flipping t within a 4-tap group is gk^3. For an even
// column pair {cp, cp+1}, the flipped pair {cp^3, cp^3-1} is the contiguous
// float2 at column cp^2, reversed.
template <int FLIP>
__global__ __launch_bounds__(256)
void vqvae_wmma_gemm_kernel(const float* __restrict__ W, const bf16* __restrict__ Bmt,
                            const float* __restrict__ bias, float* __restrict__ out,
                            int M, int N, int Ktot,
                            int lout_shift, int plain) {
    __shared__ bf16 lA[2][TM][APITCH];
    __shared__ bf16 lBt[2][TN][TK];

    const int tid  = threadIdx.x;
    const int wave = tid >> 5;
    const int lane = tid & 31;
    const int half = lane >> 4;
    const int lr   = lane & 15;
    const int m0   = blockIdx.x * TM;
    const int n0   = blockIdx.y * TN;

    // --- A staging mapping: column pair cp, rows rA + 16*ii ---
    const int cp     = (tid & 15) * 2;
    const int colEff = FLIP ? (cp ^ 2) : cp;
    const int rA     = tid >> 4;
    const float* wbase = W + (size_t)(m0 + rA) * Ktot + colEff;
    const bf16*  gbase = Bmt + (size_t)n0 * Ktot;
    const int mr = wave * 16 + lr;

    v8f acc[8];
#pragma unroll
    for (int i = 0; i < 8; ++i) acc[i] = (v8f){};

    v2f regs2[8];
    const int nsteps = Ktot / TK;

    auto loadA = [&](int k0) {
#pragma unroll
        for (int ii = 0; ii < 8; ++ii)
            regs2[ii] = *(const v2f*)(wbase + k0 + (size_t)(16 * ii) * Ktot);
    };
    auto asyncB = [&](int bufidx, int k0) {
#pragma unroll
        for (int i2 = 0; i2 < 2; ++i2) {          // 2 x 16B async chunks/thread
            int i  = tid + i2 * 256;
            int n  = i >> 2;
            int kc = (i & 3) * 8;
            unsigned lds_off = (unsigned)(uintptr_t)&lBt[bufidx][n][kc];
            asm volatile("global_load_async_to_lds_b128 %0, %1, off"
                         :: "v"(lds_off), "v"(gbase + (size_t)n * Ktot + k0 + kc) : "memory");
        }
    };
    auto commitA = [&](int bufidx) {              // pack 2 bf16 -> one b32 store
#pragma unroll
        for (int ii = 0; ii < 8; ++ii) {
            union { unsigned u; bf16 h[2]; } pk;
            pk.h[0] = (bf16)regs2[ii][FLIP ? 1 : 0];
            pk.h[1] = (bf16)regs2[ii][FLIP ? 0 : 1];
            *(unsigned*)&lA[bufidx][rA + 16 * ii][cp] = pk.u;
        }
    };

    // ---- prologue: stage tile 0 into buffer 0 ----
    loadA(0);
    asyncB(0, 0);
    commitA(0);
    asm volatile("s_wait_asynccnt 0x0" ::: "memory");
    __syncthreads();

    for (int s = 0; s < nsteps; ++s) {
        const int buf = s & 1;
        const bool has_next = (s + 1) < nsteps;

        if (has_next) {                           // next tile in flight
            loadA((s + 1) * TK);
            asyncB(1 - buf, (s + 1) * TK);
        }

        // ---- compute current tile: 1 A fragment, 8 WMMAs ----
        // A layout: dwords 0..3 -> K = half*8 + 2j; dwords 4..7 -> K = 16 + half*8 + 2j
        v16bf a;
        {
            v8bf alo = *(const v8bf*)&lA[buf][mr][half * 8];
            v8bf ahi = *(const v8bf*)&lA[buf][mr][16 + half * 8];
#pragma unroll
            for (int j = 0; j < 8; ++j) { a[j] = alo[j]; a[j + 8] = ahi[j]; }
        }
        // B layout: dwords 0..3 -> K = half*8 + 2j; dwords 4..7 -> K = 16 + half*8 + 2j; column n = lane&15
#pragma unroll
        for (int nt = 0; nt < 8; ++nt) {
            v16bf b;
            {
                v8bf blo = *(const v8bf*)&lBt[buf][nt * 16 + lr][half * 8];
                v8bf bhi = *(const v8bf*)&lBt[buf][nt * 16 + lr][16 + half * 8];
#pragma unroll
                for (int j = 0; j < 8; ++j) { b[j] = blo[j]; b[j + 8] = bhi[j]; }
            }
            acc[nt] = __builtin_amdgcn_wmma_f32_16x16x32_bf16(
                          false, a, false, b, (short)0, acc[nt], false, false);
        }

        if (has_next) {
            commitA(1 - buf);
            asm volatile("s_wait_asynccnt 0x0" ::: "memory");
        }
        __syncthreads();
    }

    // ---- epilogue: D layout row = r + 8*half, col = lane&15; fused bias ----
    float bia[8];
#pragma unroll
    for (int r = 0; r < 8; ++r) {
        int m = m0 + wave * 16 + r + 8 * half;
        bia[r] = bias ? bias[m] : 0.0f;
    }
#pragma unroll
    for (int nt = 0; nt < 8; ++nt) {
        int n = n0 + nt * 16 + lr;
        if (plain) {
#pragma unroll
            for (int r = 0; r < 8; ++r) {
                int m = m0 + wave * 16 + r + 8 * half;
                out[(size_t)m * N + n] = acc[nt][r] + bia[r];
            }
        } else {                                  // NCL layout; Lout = 1<<lout_shift
            int bb = n >> lout_shift;
            int lo = n - (bb << lout_shift);
#pragma unroll
            for (int r = 0; r < 8; ++r) {
                int m = m0 + wave * 16 + r + 8 * half;
                out[(((size_t)bb * M + m) << lout_shift) + lo] = acc[nt][r] + bia[r];
            }
        }
    }
}

// ======================= BatchNorm =========================================
__global__ void vqvae_bn_stats_kernel(const float* __restrict__ x, float* mean,
                                      float* rstd, int Bsz, int C, int L) {
    int c = blockIdx.x, tid = threadIdx.x;
    int n = Bsz * L;
    float s = 0.f, s2 = 0.f;
    for (int i = tid; i < n; i += blockDim.x) {
        int b = i / L, l = i - b * L;
        float v = x[((size_t)b * C + c) * L + l];
        s += v; s2 += v * v;
    }
    __shared__ float sh[256], sh2[256];
    sh[tid] = s; sh2[tid] = s2; __syncthreads();
    for (int st = 128; st > 0; st >>= 1) {
        if (tid < st) { sh[tid] += sh[tid + st]; sh2[tid] += sh2[tid + st]; }
        __syncthreads();
    }
    if (tid == 0) {
        float m = sh[0] / n;
        float var = sh2[0] / n - m * m;
        mean[c] = m;
        rstd[c] = rsqrtf(var + 1e-5f);
    }
}

__global__ void vqvae_bn_apply_kernel(const float* __restrict__ x, float* __restrict__ out,
                                      const float* __restrict__ mean, const float* __restrict__ rstd,
                                      const float* __restrict__ g, const float* __restrict__ bta,
                                      const float* __restrict__ residual,
                                      int C, int L, size_t total, int do_relu) {
    size_t idx = (size_t)blockIdx.x * blockDim.x + threadIdx.x;
    if (idx >= total) return;
    int c = (int)((idx / L) % C);
    float v = (x[idx] - mean[c]) * rstd[c] * g[c] + bta[c];
    if (residual) v += residual[idx];
    if (do_relu) v = fmaxf(v, 0.0f);
    out[idx] = v;
}

// ======================= elementwise =======================================
__global__ void vqvae_relu_copy_kernel(const float* __restrict__ x, float* __restrict__ out,
                                       size_t total) {
    size_t i = (size_t)blockIdx.x * blockDim.x + threadIdx.x;
    if (i < total) out[i] = fmaxf(x[i], 0.0f);
}

__global__ void vqvae_tanh_kernel(const float* __restrict__ x, float* __restrict__ out,
                                  size_t total) {
    size_t i = (size_t)blockIdx.x * blockDim.x + threadIdx.x;
    if (i < total) out[i] = tanhf(x[i]);
}

// ======================= VQ ================================================
__global__ void vqvae_rowsq_kernel(const float* __restrict__ emb, float* __restrict__ out, int D) {
    int k = blockIdx.x, tid = threadIdx.x;
    float s = 0.f;
    for (int d = tid; d < D; d += blockDim.x) {
        float v = emb[(size_t)k * D + d];
        s += v * v;
    }
    __shared__ float sh[256];
    sh[tid] = s; __syncthreads();
    for (int st = 128; st > 0; st >>= 1) {
        if (tid < st) sh[tid] += sh[tid + st];
        __syncthreads();
    }
    if (tid == 0) out[k] = sh[0];
}

__global__ void vqvae_argmin_kernel(const float* __restrict__ scores,
                                    const float* __restrict__ emb_sq,
                                    int* __restrict__ idx, int Kc, int N) {
    int pos = blockIdx.x, tid = threadIdx.x;
    float best = 3.4e38f; int bi = 0x7fffffff;
    for (int k = tid; k < Kc; k += blockDim.x) {
        float d = emb_sq[k] - 2.0f * scores[(size_t)k * N + pos];
        if (d < best || (d == best && k < bi)) { best = d; bi = k; }
    }
    __shared__ float sb[256]; __shared__ int si[256];
    sb[tid] = best; si[tid] = bi; __syncthreads();
    for (int st = 128; st > 0; st >>= 1) {
        if (tid < st) {
            if (sb[tid + st] < sb[tid] ||
                (sb[tid + st] == sb[tid] && si[tid + st] < si[tid])) {
                sb[tid] = sb[tid + st]; si[tid] = si[tid + st];
            }
        }
        __syncthreads();
    }
    if (tid == 0) idx[pos] = si[0];
}

__global__ void vqvae_gather_kernel(const float* __restrict__ emb, const int* __restrict__ idx,
                                    float* __restrict__ zq, int Bsz, int D, int Lp) {
    size_t i = (size_t)blockIdx.x * blockDim.x + threadIdx.x;
    size_t total = (size_t)Bsz * D * Lp;
    if (i >= total) return;
    int l = (int)(i % Lp);
    int d = (int)((i / Lp) % D);
    int b = (int)(i / ((size_t)Lp * D));
    zq[i] = emb[(size_t)idx[b * Lp + l] * D + d];
}

__global__ void vqvae_l2o_kernel(const float* __restrict__ zq, const float* __restrict__ w,
                                 const float* __restrict__ b0, float* __restrict__ out,
                                 int D, int Lp) {
    int n = blockIdx.x;
    int b = n / Lp, l = n - b * Lp;
    int tid = threadIdx.x;
    float s = 0.f;
    for (int d = tid; d < D; d += blockDim.x)
        s += zq[((size_t)b * D + d) * Lp + l] * w[d];
    __shared__ float sh[256];
    sh[tid] = s; __syncthreads();
    for (int st = 128; st > 0; st >>= 1) {
        if (tid < st) sh[tid] += sh[tid + st];
        __syncthreads();
    }
    if (tid == 0) out[n] = sh[0] + b0[0];
}

// ======================= host-side orchestration ===========================

namespace {

const size_t MB = 1024ull * 1024ull;

struct Ctx {
    hipStream_t stream;
    float* act0; float* act1; float* act2;
    bf16*  bmat;
    float* scores;
    float* mean; float* rstd; float* embsq;
    int*   vidx;
};

inline size_t ceil_div(size_t a, size_t b) { return (a + b - 1) / b; }
inline int ilog2(int v) { int s = 0; while ((1 << s) < v) ++s; return s; }

void run_conv(Ctx& c, const float* actin, int Bsz, int Cin, int Lin,
              const float* W, const float* bias, int Cout,
              int ktap, int stride, int pad, int lhs_dil, int flip,
              float* out, int plain) {
    int Ldil = (Lin - 1) * lhs_dil + 1;
    int Lout = (Ldil + 2 * pad - ktap) / stride + 1;
    int N = Bsz * Lout;
    int Ktot = Cin * ktap;

    size_t total = (size_t)Ktot * N;
    vqvae_im2col_kernel<<<(unsigned)ceil_div(total, 256), 256, 0, c.stream>>>(
        actin, c.bmat, Bsz, Cin, Lin, ktap, stride, pad, lhs_dil, Lout);

    dim3 grid(Cout / TM, N / TN);
    if (flip)
        vqvae_wmma_gemm_kernel<1><<<grid, 256, 0, c.stream>>>(
            W, c.bmat, bias, out, Cout, N, Ktot, ilog2(Lout), plain);
    else
        vqvae_wmma_gemm_kernel<0><<<grid, 256, 0, c.stream>>>(
            W, c.bmat, bias, out, Cout, N, Ktot, ilog2(Lout), plain);
}

void run_bn(Ctx& c, const float* x, float* out, const float* g, const float* b,
            const float* residual, int Bsz, int C, int L, int relu) {
    vqvae_bn_stats_kernel<<<C, 256, 0, c.stream>>>(x, c.mean, c.rstd, Bsz, C, L);
    size_t total = (size_t)Bsz * C * L;
    vqvae_bn_apply_kernel<<<(unsigned)ceil_div(total, 256), 256, 0, c.stream>>>(
        x, out, c.mean, c.rstd, g, b, residual, C, L, total, relu);
}

struct ResParams {
    const float *w1, *b1, *g1, *bb1, *w2, *b2, *g2, *bb2;
};

void run_resblock(Ctx& c, const float* xin, float* xout, const ResParams& p, int Lp) {
    const int Bsz = 2, D = 2048;
    size_t total = (size_t)Bsz * D * Lp;
    vqvae_relu_copy_kernel<<<(unsigned)ceil_div(total, 256), 256, 0, c.stream>>>(
        xin, c.act0, total);
    run_conv(c, c.act0, Bsz, D, Lp, p.w1, p.b1, D, 3, 1, 1, 1, 0, c.act2, 0);
    run_bn(c, c.act2, c.act2, p.g1, p.bb1, nullptr, Bsz, D, Lp, 1);
    run_conv(c, c.act2, Bsz, D, Lp, p.w2, p.b2, D, 1, 1, 0, 1, 0, c.act0, 0);
    run_bn(c, c.act0, xout, p.g2, p.bb2, xin, Bsz, D, Lp, 0);
}

} // namespace

extern "C" void kernel_launch(void* const* d_in, const int* in_sizes, int n_in,
                              void* d_out, int out_size, void* d_ws, size_t ws_size,
                              hipStream_t stream) {
    (void)in_sizes; (void)n_in; (void)out_size; (void)ws_size;

    enum {
        IN_X = 0,
        IN_ENC_W1, IN_ENC_B1, IN_ENC_G1, IN_ENC_BB1,
        IN_ENC_W2, IN_ENC_B2,
        IN_ER1, IN_ER2 = IN_ER1 + 8,
        IN_EMB = IN_ER2 + 8, IN_L2O_W, IN_L2O_B,
        IN_DR1, IN_DR2 = IN_DR1 + 8,
        IN_DEC_W1 = IN_DR2 + 8, IN_DEC_B1, IN_DEC_G1, IN_DEC_BB1,
        IN_DEC_W2, IN_DEC_B2
    };
    auto F = [&](int i) { return (const float*)d_in[i]; };
    auto RP = [&](int base) {
        ResParams p;
        p.w1 = F(base + 0); p.b1 = F(base + 1); p.g1 = F(base + 2); p.bb1 = F(base + 3);
        p.w2 = F(base + 4); p.b2 = F(base + 5); p.g2 = F(base + 6); p.bb2 = F(base + 7);
        return p;
    };

    char* ws = (char*)d_ws;
    Ctx c;
    c.stream = stream;
    c.act0   = (float*)(ws + 0 * MB);
    c.act1   = (float*)(ws + 8 * MB);
    c.act2   = (float*)(ws + 16 * MB);
    c.bmat   = (bf16*) (ws + 24 * MB);
    c.scores = (float*)(ws + 56 * MB);
    c.mean   = (float*)(ws + 57 * MB);
    c.rstd   = (float*)(ws + 57 * MB + 16384);
    c.embsq  = (float*)(ws + 57 * MB + 32768);
    c.vidx   = (int*)  (ws + 57 * MB + 40960);

    float* out_f   = (float*)d_out;
    float* xt_out  = out_f;                 // x_tilde (2,256,1024)  = 524288
    float* ze_out  = out_f + 524288;        // z_e     (2,2048,256)  = 1048576
    float* zq_out  = out_f + 1572864;       // z_q_bar (2,2048,256)  = 1048576
    float* oh_out  = out_f + 2621440;       // onehot  (2,1,256)     = 512

    // ============================ encoder ============================
    run_conv(c, F(IN_X), 2, 256, 1024, F(IN_ENC_W1), F(IN_ENC_B1),
             2048, 4, 2, 1, 1, 0, c.act0, 0);                      // L 1024->512
    run_bn(c, c.act0, c.act0, F(IN_ENC_G1), F(IN_ENC_BB1), nullptr, 2, 2048, 512, 1);
    run_conv(c, c.act0, 2, 2048, 512, F(IN_ENC_W2), F(IN_ENC_B2),
             2048, 4, 2, 1, 1, 0, c.act1, 0);                      // L 512->256
    run_resblock(c, c.act1, c.act1, RP(IN_ER1), 256);
    run_resblock(c, c.act1, ze_out, RP(IN_ER2), 256);              // z_e

    // ============================ VQ =================================
    vqvae_rowsq_kernel<<<512, 256, 0, stream>>>(F(IN_EMB), c.embsq, 2048);
    run_conv(c, ze_out, 2, 2048, 256, F(IN_EMB), nullptr,
             512, 1, 1, 0, 1, 0, c.scores, 1);
    vqvae_argmin_kernel<<<512, 256, 0, stream>>>(c.scores, c.embsq, c.vidx, 512, 512);
    {
        size_t total = 2ull * 2048 * 256;
        vqvae_gather_kernel<<<(unsigned)ceil_div(total, 256), 256, 0, stream>>>(
            F(IN_EMB), c.vidx, zq_out, 2, 2048, 256);              // z_q_bar
    }
    vqvae_l2o_kernel<<<512, 256, 0, stream>>>(zq_out, F(IN_L2O_W), F(IN_L2O_B),
                                              oh_out, 2048, 256);  // onehot

    // ============================ decoder ============================
    run_resblock(c, zq_out, c.act1, RP(IN_DR1), 256);
    run_resblock(c, c.act1, c.act1, RP(IN_DR2), 256);
    {
        size_t total = 2ull * 2048 * 256;
        vqvae_relu_copy_kernel<<<(unsigned)ceil_div(total, 256), 256, 0, stream>>>(
            c.act1, c.act0, total);
    }
    // ConvTranspose(stride2,pad1,k4) == lhs-dilated conv, pad e=2, flipped taps
    run_conv(c, c.act0, 2, 2048, 256, F(IN_DEC_W1), F(IN_DEC_B1),
             2048, 4, 1, 2, 2, 1, c.act2, 0);                      // L 256->512
    run_bn(c, c.act2, c.act2, F(IN_DEC_G1), F(IN_DEC_BB1), nullptr, 2, 2048, 512, 1);
    run_conv(c, c.act2, 2, 2048, 512, F(IN_DEC_W2), F(IN_DEC_B2),
             256, 4, 1, 2, 2, 1, c.act0, 0);                       // L 512->1024
    {
        size_t total = 2ull * 256 * 1024;
        vqvae_tanh_kernel<<<(unsigned)ceil_div(total, 256), 256, 0, stream>>>(
            c.act0, xt_out, total);                                // x_tilde
    }
}
